// LSTMModel_43069932044692
// MI455X (gfx1250) — matrix operand order repaired and checked by reading, not compiled, not run on hardware
//
#include <hip/hip_runtime.h>
#include <hip/hip_bf16.h>
#include <stdint.h>

#define BATCH   64
#define SEQT    512
#define EMBD    512
#define HID     1024
#define NG      4096      // 4*HID, gate order i,f,g,o
#define NCLS    32000
#define VOCABM1 31999
#define KCHUNK  128
#define NCHUNK  (HID / KCHUNK)   // 8

typedef __attribute__((ext_vector_type(16))) __bf16 v16bf;
typedef __attribute__((ext_vector_type(8)))  __bf16 v8bf;
typedef __attribute__((ext_vector_type(8)))  float  v8f;

typedef __attribute__((ext_vector_type(4))) unsigned int u32x4;
typedef __attribute__((ext_vector_type(4))) int          i32x4;
typedef __attribute__((ext_vector_type(8))) int          i32x8;

#if defined(__AMDGCN__) && __has_builtin(__builtin_amdgcn_tensor_load_to_lds) && \
    __has_builtin(__builtin_amdgcn_s_wait_tensorcnt)
#define USE_TDM 1
#else
#define USE_TDM 0
#endif

static __device__ __forceinline__ v8f wmma_bf16(v16bf a, v16bf b, v8f c) {
  // D = A(16x32 bf16) * B(32x16 bf16) + C(16x16 f32)
  return __builtin_amdgcn_wmma_f32_16x16x32_bf16(false, a, false, b, (short)0, c,
                                                 false, false);
}

// A fragment 16x32 (MxK), source row-major with K contiguous, row stride ld.
// Layout: lanes 0-15 hold row M=lane, K 0-15;
// lanes 16-31 hold the same rows with K 16-31 (matches frag_b).
static __device__ __forceinline__ v16bf frag_a(const __bf16* base, int ld, int lane) {
  const int r  = lane & 15;
  const int kb = (lane >> 4) << 4;   // 0 or 16
  const __bf16* p = base + r * ld + kb;
  v8bf lo = *(const v8bf*)(p);
  v8bf hi = *(const v8bf*)(p + 8);
  v16bf v;
#pragma unroll
  for (int i = 0; i < 8; ++i) { v[i] = lo[i]; v[i + 8] = hi[i]; }
  return v;
}

// B fragment 32x16 (KxN): lanes 0-15: col=lane, K 0-15; lanes 16-31: col=lane-16,
// K 16-31 -> 32 contiguous bytes per lane from weight row (= column of B).
static __device__ __forceinline__ v16bf frag_b(const __bf16* base, int ld, int lane) {
  const int c  = lane & 15;
  const int kb = (lane >> 4) << 4;   // 0 or 16
  return *(const v16bf*)(base + c * ld + kb);
}

static __device__ __forceinline__ float sigmoidf_(float x) {
  return 1.0f / (1.0f + __expf(-x));
}

#if USE_TDM
// 32-bit LDS byte offset of a __shared__ object (ptrtoint of an AS(3) pointer).
static __device__ __forceinline__ unsigned lds_off_u32(const void* p) {
  return (unsigned)(unsigned long long)(const __attribute__((address_space(3))) void*)p;
}

// Issue a TDM 2-D tile load: rows x cols bf16 elements from a row-major tensor
// (row stride = rowstride elements) into contiguous LDS at ldsoff.
static __device__ __forceinline__ void tdm_load_2d(const __bf16* gsrc, unsigned ldsoff,
                                                   int rows, int cols, int rowstride) {
  const unsigned long long ga = (unsigned long long)(size_t)gsrc;
  u32x4 g0;
  g0[0] = 1u;                                       // count=1, user descriptor
  g0[1] = ldsoff;                                   // lds_addr (bytes)
  g0[2] = (unsigned)(ga & 0xFFFFFFFFu);             // global_addr[31:0]
  g0[3] = (unsigned)((ga >> 32) & 0x1FFFFFFu)       // global_addr[56:32]
        | (2u << 30);                               // type=2 ("image")
  i32x8 g1;
  g1[0] = (int)(1u << 16);                          // wg_mask=0, data_size=1 (2B)
  g1[1] = (int)((unsigned)rowstride << 16);         // tensor_dim0[15:0] @ b63:48
  g1[2] = (int)((((unsigned)rowstride >> 16) & 0xFFFFu) | ((unsigned)rows << 16));
  g1[3] = (int)((((unsigned)rows >> 16) & 0xFFFFu) | ((unsigned)cols << 16)); // tile_dim0
  g1[4] = rows;                                     // tile_dim1, tile_dim2=0
  g1[5] = rowstride;                                // tensor_dim0_stride[31:0]
  g1[6] = 0;
  g1[7] = 0;
  const i32x4 z4 = {0, 0, 0, 0};
#if __clang_major__ >= 23
  const i32x8 z8 = {0, 0, 0, 0, 0, 0, 0, 0};
  __builtin_amdgcn_tensor_load_to_lds(g0, g1, z4, z4, z8, 0);
#else
  __builtin_amdgcn_tensor_load_to_lds(g0, g1, z4, z4, 0);
#endif
}
#endif  // USE_TDM

// ---------------------------------------------------------------------------
// Pre-pass kernels
// ---------------------------------------------------------------------------
__global__ void f2bf_kernel(const float* __restrict__ s, __bf16* __restrict__ d, size_t n) {
  for (size_t i = (size_t)blockIdx.x * blockDim.x + threadIdx.x; i < n;
       i += (size_t)gridDim.x * blockDim.x)
    d[i] = (__bf16)s[i];
}

__global__ void bias_combine_kernel(const float* __restrict__ a, const float* __restrict__ b,
                                    float* __restrict__ o, int n) {
  int i = blockIdx.x * blockDim.x + threadIdx.x;
  if (i < n) o[i] = a[i] + b[i];
}

__global__ void fill_bf16_kernel(__bf16* __restrict__ p, size_t n) {
  for (size_t i = (size_t)blockIdx.x * blockDim.x + threadIdx.x; i < n;
       i += (size_t)gridDim.x * blockDim.x)
    p[i] = (__bf16)0.0f;
}

__global__ void fill_f32_kernel(float* __restrict__ p, size_t n) {
  for (size_t i = (size_t)blockIdx.x * blockDim.x + threadIdx.x; i < n;
       i += (size_t)gridDim.x * blockDim.x)
    p[i] = 0.0f;
}

__global__ void embed_kernel(const int* __restrict__ feat, const float* __restrict__ emb,
                             __bf16* __restrict__ x) {
  const int bt = blockIdx.x;                 // b*SEQT + t
  const int tok = feat[bt];
  const float* e = emb + (size_t)tok * EMBD;
  __bf16* xo = x + (size_t)bt * EMBD;
  for (int i = threadIdx.x; i < EMBD; i += blockDim.x) xo[i] = (__bf16)e[i];
}

__global__ void seqlen_kernel(const int* __restrict__ feat, int* __restrict__ seq) {
  const int b = threadIdx.x;
  if (b >= BATCH) return;
  int cnt = 0;
  for (int t = 0; t < SEQT; ++t) cnt += (feat[b * SEQT + t] != VOCABM1) ? 1 : 0;
  seq[b] = cnt - 1;
}

// ---------------------------------------------------------------------------
// Generic bf16 TN GEMM:  C[M,N] = A[M,K] * W[N,K]^T (+ bias[N])
// Block = 64x64 tile, 256 threads = 8 wave32, wave -> (row tile, 2 col tiles).
// permuteBT=1: output row m=(b*SEQT+t) written to row (t*BATCH+b), bf16.
// permuteBT=0: f32 output, identity row mapping.
// ---------------------------------------------------------------------------
__global__ void __launch_bounds__(256)
gemm_bf16_tn(const __bf16* __restrict__ A, int lda,
             const __bf16* __restrict__ W, int ldw,
             const float* __restrict__ bias,
             float* __restrict__ outF, __bf16* __restrict__ outB,
             int N, int K, int permuteBT)
{
  const int lane = threadIdx.x & 31;
  const int w    = threadIdx.x >> 5;
  const int rt   = w & 3;              // 4 row tiles of 16
  const int ctp  = (w >> 2) << 1;      // col tile pair: {0,1} or {2,3}
  const int mrow = blockIdx.y * 64 + rt * 16;
  const int ncol = blockIdx.x * 64;

  v8f acc0 = {0.f,0.f,0.f,0.f,0.f,0.f,0.f,0.f};
  v8f acc1 = acc0;

  const __bf16* aptr = A + (size_t)mrow * lda;
  const __bf16* b0p  = W + (size_t)(ncol + ctp * 16) * ldw;
  const __bf16* b1p  = b0p + (size_t)16 * ldw;

  for (int k = 0; k < K; k += 32) {
    v16bf a  = frag_a(aptr + k, lda, lane);
    v16bf b0 = frag_b(b0p + k, ldw, lane);
    v16bf b1 = frag_b(b1p + k, ldw, lane);
    acc0 = wmma_bf16(a, b0, acc0);
    acc1 = wmma_bf16(a, b1, acc1);
  }

  const int lc = lane & 15;
  const int rb = (lane >> 4) << 3;     // row offset 0 / 8 inside tile
#pragma unroll
  for (int cse = 0; cse < 2; ++cse) {
    const v8f acc = cse ? acc1 : acc0;
    const int n = ncol + (ctp + cse) * 16 + lc;
    const float bv = bias ? bias[n] : 0.0f;
#pragma unroll
    for (int r = 0; r < 8; ++r) {
      const int m = mrow + rb + r;
      const float v = acc[r] + bv;
      if (permuteBT) {
        const int bb = m >> 9;               // m / SEQT
        const int tt = m & (SEQT - 1);       // m % SEQT
        outB[((size_t)tt * BATCH + bb) * N + n] = (__bf16)v;
      } else {
        outF[(size_t)m * N + n] = v;
      }
    }
  }
}

// ---------------------------------------------------------------------------
// Shared WMMA micro-kernel for one K-chunk of the recurrent GEMM.
// ---------------------------------------------------------------------------
static __device__ __forceinline__ void chunk_mma(const __bf16* hb, const __bf16* __restrict__ Whh,
                                                 int jb, int kc, int rh, int lane,
                                                 v8f acc[2][4]) {
#pragma unroll
  for (int kk = 0; kk < KCHUNK; kk += 32) {
    const v16bf a0 = frag_a(hb + (rh * 32 + 0)  * KCHUNK + kk, KCHUNK, lane);
    const v16bf a1 = frag_a(hb + (rh * 32 + 16) * KCHUNK + kk, KCHUNK, lane);
#pragma unroll
    for (int g = 0; g < 4; ++g) {
      const v16bf b = frag_b(Whh + (size_t)(g * HID + jb) * HID + kc + kk, HID, lane);
      acc[0][g] = wmma_bf16(a0, b, acc[0][g]);
      acc[1][g] = wmma_bf16(a1, b, acc[1][g]);
    }
  }
}

// ---------------------------------------------------------------------------
// One LSTM timestep: gates[64,4H] = xg_t + h_prev @ W_hh^T, then cell update.
// Grid = HID/64 blocks, 256 threads (8 waves). Each wave owns 16 h-columns and
// computes all 4 gates for 32 batch rows -> pointwise update fully in-register.
// h_prev is staged into LDS via the Tensor Data Mover (double-buffered, DMA
// overlapped with WMMA); W_hh (bf16, L2-resident) streamed as B fragments.
// ---------------------------------------------------------------------------
__global__ void __launch_bounds__(256)
lstm_step_kernel(const __bf16* __restrict__ xg,    // [SEQT, BATCH, NG] bf16
                 const __bf16* __restrict__ Whh,   // [NG, HID] bf16
                 const __bf16* __restrict__ h_in,  // [BATCH, HID] bf16
                 float*        __restrict__ c_st,  // [BATCH, HID] f32 (in-place)
                 __bf16*       __restrict__ h_out, // [BATCH, HID] bf16
                 __bf16*       __restrict__ h_seq, // [BATCH, SEQT, HID] or null
                 const int*    __restrict__ seq_lens,
                 __bf16*       __restrict__ last,  // [BATCH, HID] or null
                 int t)
{
  __shared__ __align__(64) __bf16 hS[2][64 * KCHUNK];   // 2 x 16 KB double buffer

  const int tid  = threadIdx.x;
  const int lane = tid & 31;
  const int w    = tid >> 5;
  const int ct   = w & 3;                 // 16-col tile within 64-col block
  const int rh   = w >> 2;                // 0: rows 0-31, 1: rows 32-63
  const int jb   = blockIdx.x * 64 + ct * 16;

  const __bf16* xgt = xg + (size_t)t * BATCH * NG;
  __builtin_prefetch(xgt + (size_t)(rh * 32) * NG + jb, 0, 0);

  const v8f zero = {0.f,0.f,0.f,0.f,0.f,0.f,0.f,0.f};
  v8f acc[2][4];
#pragma unroll
  for (int i = 0; i < 2; ++i)
#pragma unroll
    for (int g = 0; g < 4; ++g) acc[i][g] = zero;

#if USE_TDM
  // TDM path: wave 0 drives the DMA engine (TDM ignores EXEC -> PC-guard the
  // issue per-wave); all waves consume. TENSORcnt tracks completion.
  const bool issuer = (tid < 32);
  if (issuer) tdm_load_2d(h_in, lds_off_u32(&hS[0][0]), 64, KCHUNK, HID);
  for (int ic = 0; ic < NCHUNK; ++ic) {
    if (issuer) __builtin_amdgcn_s_wait_tensorcnt(0);   // chunk ic landed in LDS
    __syncthreads();                                    // publish to all waves
    if (issuer && (ic + 1 < NCHUNK))                    // prefetch chunk ic+1
      tdm_load_2d(h_in + (ic + 1) * KCHUNK, lds_off_u32(&hS[(ic + 1) & 1][0]),
                  64, KCHUNK, HID);
    chunk_mma(&hS[ic & 1][0], Whh, jb, ic * KCHUNK, rh, lane, acc);
  }
#else
  // Fallback: manual cooperative staging through VGPRs.
  for (int kc = 0; kc < HID; kc += KCHUNK) {
#pragma unroll
    for (int i = tid; i < 64 * KCHUNK / 8; i += 256) {
      const int row = i / (KCHUNK / 8);
      const int col = (i - row * (KCHUNK / 8)) * 8;
      *(v8bf*)(&hS[0][0] + row * KCHUNK + col) =
          *(const v8bf*)(h_in + (size_t)row * HID + kc + col);
    }
    __syncthreads();
    chunk_mma(&hS[0][0], Whh, jb, kc, rh, lane, acc);
    __syncthreads();
  }
#endif

  // pointwise cell update (gate order i, f, g, o)
  const int lc = lane & 15;
  const int rb = (lane >> 4) << 3;
#pragma unroll
  for (int rt2 = 0; rt2 < 2; ++rt2) {
#pragma unroll
    for (int r = 0; r < 8; ++r) {
      const int m = rh * 32 + rt2 * 16 + rb + r;
      const int n = jb + lc;
      const size_t xrow = (size_t)m * NG;
      float gi = acc[rt2][0][r] + (float)xgt[xrow + 0 * HID + n];
      float gf = acc[rt2][1][r] + (float)xgt[xrow + 1 * HID + n];
      float gg = acc[rt2][2][r] + (float)xgt[xrow + 2 * HID + n];
      float go = acc[rt2][3][r] + (float)xgt[xrow + 3 * HID + n];
      gi = sigmoidf_(gi);
      gf = sigmoidf_(gf);
      gg = tanhf(gg);
      go = sigmoidf_(go);
      const size_t idx = (size_t)m * HID + n;
      const float c = gf * c_st[idx] + gi * gg;
      c_st[idx] = c;
      const float h = go * tanhf(c);
      const __bf16 hb = (__bf16)h;
      h_out[idx] = hb;
      if (h_seq) h_seq[((size_t)m * SEQT + t) * HID + n] = hb;
      if (last && seq_lens[m] == t) last[idx] = hb;
    }
  }
}

// ---------------------------------------------------------------------------
// Host launch
// ---------------------------------------------------------------------------
extern "C" void kernel_launch(void* const* d_in, const int* in_sizes, int n_in,
                              void* d_out, int out_size, void* d_ws, size_t ws_size,
                              hipStream_t stream)
{
  (void)in_sizes; (void)n_in; (void)out_size; (void)ws_size;

  const int*   feat = (const int*)d_in[0];
  const float* emb  = (const float*)d_in[1];
  const float* wih0 = (const float*)d_in[2];
  const float* whh0 = (const float*)d_in[3];
  const float* bih0 = (const float*)d_in[4];
  const float* bhh0 = (const float*)d_in[5];
  const float* wih1 = (const float*)d_in[6];
  const float* whh1 = (const float*)d_in[7];
  const float* bih1 = (const float*)d_in[8];
  const float* bhh1 = (const float*)d_in[9];
  const float* fcw  = (const float*)d_in[10];
  const float* fcb  = (const float*)d_in[11];
  float* out = (float*)d_out;

  uint8_t* ws = (uint8_t*)d_ws;
  size_t off = 0;
  auto alloc = [&](size_t bytes) -> uint8_t* {
    uint8_t* p = ws + off;
    off += (bytes + 255) & ~(size_t)255;
    return p;
  };

  __bf16* wih0_b = (__bf16*)alloc((size_t)NG * EMBD * 2);
  __bf16* whh0_b = (__bf16*)alloc((size_t)NG * HID * 2);
  __bf16* wih1_b = (__bf16*)alloc((size_t)NG * HID * 2);
  __bf16* whh1_b = (__bf16*)alloc((size_t)NG * HID * 2);
  __bf16* fcw_b  = (__bf16*)alloc((size_t)NCLS * HID * 2);
  float*  bias0  = (float*)alloc((size_t)NG * 4);
  float*  bias1  = (float*)alloc((size_t)NG * 4);
  __bf16* x_b    = (__bf16*)alloc((size_t)BATCH * SEQT * EMBD * 2);
  __bf16* h1_b   = (__bf16*)alloc((size_t)BATCH * SEQT * HID * 2);
  __bf16* xg_b   = (__bf16*)alloc((size_t)SEQT * BATCH * NG * 2);  // reused by both layers
  __bf16* hA     = (__bf16*)alloc((size_t)BATCH * HID * 2);
  __bf16* hB     = (__bf16*)alloc((size_t)BATCH * HID * 2);
  float*  c_st   = (float*)alloc((size_t)BATCH * HID * 4);
  __bf16* last_b = (__bf16*)alloc((size_t)BATCH * HID * 2);
  int*    seql   = (int*)alloc((size_t)BATCH * 4);

  // --- weight / bias prep (bf16, L2-resident working set) ---
  f2bf_kernel<<<2048, 256, 0, stream>>>(wih0, wih0_b, (size_t)NG * EMBD);
  f2bf_kernel<<<2048, 256, 0, stream>>>(whh0, whh0_b, (size_t)NG * HID);
  f2bf_kernel<<<2048, 256, 0, stream>>>(wih1, wih1_b, (size_t)NG * HID);
  f2bf_kernel<<<2048, 256, 0, stream>>>(whh1, whh1_b, (size_t)NG * HID);
  f2bf_kernel<<<4096, 256, 0, stream>>>(fcw,  fcw_b,  (size_t)NCLS * HID);
  bias_combine_kernel<<<NG / 256, 256, 0, stream>>>(bih0, bhh0, bias0, NG);
  bias_combine_kernel<<<NG / 256, 256, 0, stream>>>(bih1, bhh1, bias1, NG);

  // --- embedding gather + sequence lengths ---
  embed_kernel<<<BATCH * SEQT, 128, 0, stream>>>(feat, emb, x_b);
  seqlen_kernel<<<1, 64, 0, stream>>>(feat, seql);

  const dim3 gXG(NG / 64, (BATCH * SEQT) / 64);

  // ===================== layer 0 =====================
  gemm_bf16_tn<<<gXG, 256, 0, stream>>>(x_b, EMBD, wih0_b, EMBD, bias0,
                                        nullptr, xg_b, NG, EMBD, 1);
  fill_bf16_kernel<<<64, 256, 0, stream>>>(hA, (size_t)BATCH * HID);
  fill_bf16_kernel<<<64, 256, 0, stream>>>(hB, (size_t)BATCH * HID);
  fill_f32_kernel<<<64, 256, 0, stream>>>(c_st, (size_t)BATCH * HID);

  {
    __bf16* hin = hA; __bf16* hout = hB;
    for (int t = 0; t < SEQT; ++t) {
      lstm_step_kernel<<<HID / 64, 256, 0, stream>>>(xg_b, whh0_b, hin, c_st, hout,
                                                     h1_b, nullptr, nullptr, t);
      __bf16* tmp = hin; hin = hout; hout = tmp;
    }
  }

  // ===================== layer 1 =====================
  gemm_bf16_tn<<<gXG, 256, 0, stream>>>(h1_b, HID, wih1_b, HID, bias1,
                                        nullptr, xg_b, NG, HID, 1);
  fill_bf16_kernel<<<64, 256, 0, stream>>>(hA, (size_t)BATCH * HID);
  fill_bf16_kernel<<<64, 256, 0, stream>>>(hB, (size_t)BATCH * HID);
  fill_f32_kernel<<<64, 256, 0, stream>>>(c_st, (size_t)BATCH * HID);
  fill_bf16_kernel<<<64, 256, 0, stream>>>(last_b, (size_t)BATCH * HID);

  {
    __bf16* hin = hA; __bf16* hout = hB;
    for (int t = 0; t < SEQT; ++t) {
      lstm_step_kernel<<<HID / 64, 256, 0, stream>>>(xg_b, whh1_b, hin, c_st, hout,
                                                     nullptr, seql, last_b, t);
      __bf16* tmp = hin; hin = hout; hout = tmp;
    }
  }

  // ===================== FC head =====================
  const dim3 gFC(NCLS / 64, (BATCH) / 64);
  gemm_bf16_tn<<<gFC, 256, 0, stream>>>(last_b, HID, fcw_b, HID, fcb,
                                        out, nullptr, NCLS, HID, 0);
}
